// RNNLM_35974646071614
// MI455X (gfx1250) — matrix operand order repaired and checked by reading, not compiled, not run on hardware
//
#include <hip/hip_runtime.h>
#include <hip/hip_bf16.h>

// ---------- CDNA5 wave32 WMMA / TDM types ----------
typedef __attribute__((ext_vector_type(16))) __bf16 v16bf;
typedef __attribute__((ext_vector_type(8)))  __bf16 v8bf;
typedef __attribute__((ext_vector_type(8)))  float  v8f;
typedef unsigned int __attribute__((ext_vector_type(4))) u32x4;
typedef int          __attribute__((ext_vector_type(8))) i32x8;
typedef int          __attribute__((ext_vector_type(4))) i32x4;

#define BSZ   32
#define TLEN  512
#define DEMB  512
#define HDIM  1024
#define G4    4096     // 4*H
#define PDIM  512

// ============================================================
// Embedding gather + fp32 -> bf16 conversion.
// ============================================================
__global__ __launch_bounds__(256) void embed_gather(
    const int* __restrict__ tok, const float* __restrict__ emb,
    __bf16* __restrict__ out) {
  size_t i = (size_t)blockIdx.x * 256 + threadIdx.x;   // over B*T*DEMB
  int row = (int)(i >> 9);                              // / 512
  int d   = (int)(i & 511);
  int tk  = tok[row];
  out[i] = (__bf16)emb[(size_t)tk * DEMB + d];
}

// ============================================================
// Weight convert + transpose: Wt[c*R + r] = bf16(W[r*C + c])
// ============================================================
__global__ __launch_bounds__(256) void conv_transpose(
    const float* __restrict__ W, __bf16* __restrict__ Wt, int R, int C) {
  size_t i = (size_t)blockIdx.x * 256 + threadIdx.x;
  if (i >= (size_t)R * C) return;
  int r = (int)(i / C);
  int c = (int)(i % C);
  Wt[(size_t)c * R + r] = (__bf16)W[i];
}

// ============================================================
// Fragment helpers (CDNA5 16x16x32 bf16 layouts, wave32)
// ============================================================
__device__ __forceinline__ v16bf load_a_frag(const __bf16* rowp, int k, int ahalf) {
  v8bf lo = *(const v8bf*)(rowp + k + 2 * ahalf);
  v8bf hi = *(const v8bf*)(rowp + k + 8 + 2 * ahalf);
  return __builtin_shufflevector(lo, hi, 0,1,2,3,4,5,6,7,8,9,10,11,12,13,14,15);
}

// ============================================================
// Tensor Data Mover: 2D tile (rows x rowlen bf16, row stride in
// elements) from global -> LDS.  D# per CDNA5 ISA ch.8:
//  group0: [1:0]=count=1, [63:32]=lds_addr, [120:64]=global_addr,
//          [127:126]=type=2
//  group1: [17:16]=data_size(1->2B), [79:48]=tensor_dim0,
//          [111:80]=tensor_dim1, [127:112]=tile_dim0,
//          [143:128]=tile_dim1, [207:160]=tensor_dim0_stride
// (this toolchain exposes the 6-arg builtin: g0,g1,g2,g3,g4,cpol)
// ============================================================
__device__ __forceinline__ void tdm_load_2d_to_lds(
    const void* gsrc, unsigned lds_byte_addr,
    unsigned rows, unsigned rowlen, unsigned row_stride) {
  unsigned long long ga = (unsigned long long)(uintptr_t)gsrc;
  u32x4 g0;
  g0.x = 1u;                                               // count=1, user desc
  g0.y = lds_byte_addr;                                    // lds_addr
  g0.z = (unsigned)(ga & 0xFFFFFFFFu);                     // global_addr[31:0]
  g0.w = (unsigned)((ga >> 32) & 0x01FFFFFFu) | (2u << 30); // ga[56:32] | type=2
  i32x8 g1;
  g1[0] = (int)(1u << 16);                                 // data_size=1 (2 bytes)
  g1[1] = (int)((rowlen & 0xFFFFu) << 16);                 // tensor_dim0 lo16
  g1[2] = (int)((rowlen >> 16) & 0xFFFFu) | (int)((rows & 0xFFFFu) << 16); // dim0 hi | dim1 lo
  g1[3] = (int)((rows >> 16) & 0xFFFFu)   | (int)((rowlen & 0xFFFFu) << 16); // dim1 hi | tile_dim0
  g1[4] = (int)(rows & 0xFFFFu);                           // tile_dim1 (tile_dim2=0)
  g1[5] = (int)row_stride;                                 // tensor_dim0_stride[31:0]
  g1[6] = 0;                                               // stride hi | dim1_stride lo
  g1[7] = 0;
  i32x4 gz4 = {0, 0, 0, 0};
  i32x8 gz8 = {0, 0, 0, 0, 0, 0, 0, 0};
  __builtin_amdgcn_tensor_load_to_lds(g0, g1, gz4, gz4, gz8, 0);
}

// ============================================================
// Generic bf16 WMMA GEMM:  C(MxN) = A(MxK) @ Bt(NxK)^T  (+bias)
// block = 256 threads = 8 waves; wave tile 16x16; block tile 64x32.
// ============================================================
__global__ __launch_bounds__(256) void gemm_bf16(
    const __bf16* __restrict__ A, const __bf16* __restrict__ Bt,
    const float* __restrict__ bias,
    float* __restrict__ Cf, __bf16* __restrict__ Cb,
    int M, int N, int K) {
  const int lane = threadIdx.x & 31;
  const int w    = threadIdx.x >> 5;     // 0..7
  const int mt   = w & 3;                // 4 M-tiles
  const int nt   = w >> 2;               // 2 N-tiles
  const int m0 = blockIdx.y * 64 + mt * 16;
  const int n0 = blockIdx.x * 32 + nt * 16;
  const int row = m0 + (lane & 15);
  const int col = n0 + (lane & 15);
  const int ahalf = (lane >> 4) << 3;    // 0 / 8
  const int bhalf = (lane >> 4) << 4;    // 0 / 16

  const __bf16* arow = A  + (size_t)row * K;
  const __bf16* bcol = Bt + (size_t)col * K;

  v8f acc = {};
  for (int k = 0; k < K; k += 32) {
    v16bf af = load_a_frag(arow, k, ahalf);
    v16bf bf = *(const v16bf*)(bcol + k + bhalf);
    acc = __builtin_amdgcn_wmma_f32_16x16x32_bf16(
        false, af, false, bf, (short)0, acc, false, false);
  }

  const float bv = bias ? bias[col] : 0.0f;
  const int mb = m0 + ((lane >> 4) << 3);
#pragma unroll
  for (int i = 0; i < 8; ++i) {
    float v = acc[i] + bv;
    size_t off = (size_t)(mb + i) * N + col;
    if (Cf) Cf[off] = v;
    if (Cb) Cb[off] = (__bf16)v;
  }
}

// ============================================================
// Grid-wide sync (persistent kernel): monotonic arrive counter.
// ============================================================
__device__ __forceinline__ void grid_sync(unsigned* cnt, unsigned gen) {
  __syncthreads();
  if (threadIdx.x == 0) {
    __threadfence();                       // release h-writes
    atomicAdd(cnt, 1u);
    const unsigned target = gen * gridDim.x;
    while (__hip_atomic_load(cnt, __ATOMIC_ACQUIRE, __HIP_MEMORY_SCOPE_AGENT) < target)
      __builtin_amdgcn_s_sleep(2);
  }
  __syncthreads();
}

// ============================================================
// Persistent LSTM recurrence. 64 blocks; block g owns hidden
// units [g*16, g*16+16). Wh slice (64 cols x 1024 k bf16 = 128KB)
// staged into LDS by the Tensor Data Mover (4 x 2D tiles, one per
// gate), cell state lives in LDS, one grid sync per time step,
// h ping-pongs through a 64KB L2-resident buffer.
// ============================================================
__global__ __launch_bounds__(256) void lstm_scan(
    const __bf16* __restrict__ xz,    // (B*T, 4096) bf16 (X@Wx, no bias)
    const float*  __restrict__ bias,  // (4096,) fp32
    const __bf16* __restrict__ Wht,   // (4096, 1024) bf16 (Wh transposed)
    __bf16* __restrict__ ybf,         // (B*T, 1024) bf16 out (h per step)
    __bf16* __restrict__ h0buf,       // (32, 1024) ping
    __bf16* __restrict__ h1buf,       // (32, 1024) pong
    unsigned* __restrict__ sync) {
  extern __shared__ char smem[];
  __bf16* Wlds = (__bf16*)smem;                              // 64*1024 bf16
  float*  zbuf = (float*)(smem + (size_t)64 * HDIM * 2);     // 32*64 f32
  float*  cst  = zbuf + 32 * 64;                             // 32*16 f32

  const int tid = threadIdx.x;
  const int blk = blockIdx.x;       // hidden-unit group 0..63

  // --- TDM: stage Wh slice into LDS. Local col c = gate*16+j is
  //     transposed-global row gate*1024 + blk*16 + j; each gate's 16
  //     rows are contiguous, so 4 descriptors of (16 x 1024) bf16.
  if (tid < 32) {                   // wave 0 issues the DMAs
    const unsigned ldsbase = (unsigned)(uintptr_t)Wlds;
#pragma unroll
    for (int gate = 0; gate < 4; ++gate) {
      const __bf16* src = Wht + (size_t)(gate * HDIM + blk * 16) * HDIM;
      tdm_load_2d_to_lds(src, ldsbase + (unsigned)gate * 16u * HDIM * 2u,
                         16u, (unsigned)HDIM, (unsigned)HDIM);
    }
    __builtin_amdgcn_s_wait_tensorcnt(0);
  }
  // --- Init cell state and our slice of h_0 = 0.
  for (int i = tid; i < 32 * 16; i += 256) {
    cst[i] = 0.0f;
    int b = i >> 4, j = i & 15;
    h0buf[(size_t)b * HDIM + blk * 16 + j] = (__bf16)0.0f;
  }
  unsigned gen = 1;
  grid_sync(sync, gen);             // TDM done (wave0), h_0 slices visible

  // Wave tiling: 8 waves = 2 M-tiles (batch 0-15 / 16-31) x 4 N-tiles.
  const int lane  = tid & 31;
  const int w     = tid >> 5;
  const int mt    = w & 1;
  const int nt    = w >> 1;
  const int ahalf = (lane >> 4) << 3;
  const int bhalf = (lane >> 4) << 4;
  const int arow  = mt * 16 + (lane & 15);   // batch row
  const int bcol  = nt * 16 + (lane & 15);   // local gate-col
  const __bf16* bcolp = Wlds + (size_t)bcol * HDIM;

  for (int t = 0; t < TLEN; ++t) {
    const __bf16* hread  = (t & 1) ? h1buf : h0buf;
    __bf16*       hwrite = (t & 1) ? h0buf : h1buf;

    // --- z-tile = h_{t-1} @ Wh_slice  (K = 1024, 32 WMMAs/wave)
    const __bf16* arowp = hread + (size_t)arow * HDIM;
    v8f acc = {};
    for (int k = 0; k < HDIM; k += 32) {
      v16bf af = load_a_frag(arowp, k, ahalf);
      v16bf bf = *(const v16bf*)(bcolp + k + bhalf);
      acc = __builtin_amdgcn_wmma_f32_16x16x32_bf16(
          false, af, false, bf, (short)0, acc, false, false);
    }
    const int mb = mt * 16 + ((lane >> 4) << 3);
    const int nn = nt * 16 + (lane & 15);
#pragma unroll
    for (int i = 0; i < 8; ++i) zbuf[(mb + i) * 64 + nn] = acc[i];
    __syncthreads();

    // --- Gates + state update: 32 batch x 16 hidden = 512 items.
#pragma unroll
    for (int r = 0; r < 2; ++r) {
      const int idx = tid + r * 256;       // = b*16 + j
      const int b = idx >> 4, j = idx & 15;
      const int colI = blk * 16 + j;
      const size_t row = (size_t)b * TLEN + t;
      const __bf16* xzr = xz + row * G4;
      if (t + 1 < TLEN)                    // warm L2->WGP for next step
        __builtin_prefetch(xzr + G4 + colI, 0, 1);
      float zi = zbuf[b * 64 +  0 + j] + (float)xzr[0 * HDIM + colI] + bias[0 * HDIM + colI];
      float zf = zbuf[b * 64 + 16 + j] + (float)xzr[1 * HDIM + colI] + bias[1 * HDIM + colI];
      float zg = zbuf[b * 64 + 32 + j] + (float)xzr[2 * HDIM + colI] + bias[2 * HDIM + colI];
      float zo = zbuf[b * 64 + 48 + j] + (float)xzr[3 * HDIM + colI] + bias[3 * HDIM + colI];
      float iv = 1.0f / (1.0f + __expf(-zi));
      float fv = 1.0f / (1.0f + __expf(-zf));
      float gv = tanhf(zg);
      float ov = 1.0f / (1.0f + __expf(-zo));
      float cc = fv * cst[idx] + iv * gv;
      float hh = ov * tanhf(cc);
      cst[idx] = cc;
      hwrite[(size_t)b * HDIM + colI] = (__bf16)hh;
      ybf[row * HDIM + colI]          = (__bf16)hh;
    }
    grid_sync(sync, ++gen);
  }
}

// ============================================================
// Host orchestration (graph-capture safe: kernels + memsetAsync).
// ============================================================
extern "C" void kernel_launch(void* const* d_in, const int* in_sizes, int n_in,
                              void* d_out, int out_size, void* d_ws, size_t ws_size,
                              hipStream_t stream) {
  (void)in_sizes; (void)n_in; (void)out_size; (void)ws_size;
  const int*   tokens = (const int*)  d_in[0];
  const float* embed  = (const float*)d_in[1];
  const float* Wx0 = (const float*)d_in[2];
  const float* Wh0 = (const float*)d_in[3];
  const float* b0  = (const float*)d_in[4];
  const float* Wfc0= (const float*)d_in[5];
  const float* bfc0= (const float*)d_in[6];
  const float* Wx1 = (const float*)d_in[7];
  const float* Wh1 = (const float*)d_in[8];
  const float* b1  = (const float*)d_in[9];
  const float* Wfc1= (const float*)d_in[10];
  const float* bfc1= (const float*)d_in[11];

  char* ws = (char*)d_ws;
  size_t off = 0;
  auto carve = [&](size_t bytes) -> char* {
    char* p = ws + off;
    off += (bytes + 255) & ~(size_t)255;
    return p;
  };
  const size_t MB_ = (size_t)BSZ * TLEN;                  // 16384 rows
  __bf16* Xemb = (__bf16*)carve(MB_ * DEMB * 2);          // 16 MB
  __bf16* X1   = (__bf16*)carve(MB_ * PDIM * 2);          // 16 MB
  __bf16* XZ   = (__bf16*)carve(MB_ * G4   * 2);          // 128 MB (reused)
  __bf16* Y    = (__bf16*)carve(MB_ * HDIM * 2);          // 32 MB (reused)
  __bf16* Wx0t = (__bf16*)carve((size_t)G4 * DEMB * 2);
  __bf16* Wh0t = (__bf16*)carve((size_t)G4 * HDIM * 2);
  __bf16* Wf0t = (__bf16*)carve((size_t)PDIM * HDIM * 2);
  __bf16* Wx1t = (__bf16*)carve((size_t)G4 * PDIM * 2);
  __bf16* Wh1t = (__bf16*)carve((size_t)G4 * HDIM * 2);
  __bf16* Wf1t = (__bf16*)carve((size_t)PDIM * HDIM * 2);
  __bf16* h0   = (__bf16*)carve((size_t)BSZ * HDIM * 2);
  __bf16* h1   = (__bf16*)carve((size_t)BSZ * HDIM * 2);
  unsigned* syncc = (unsigned*)carve(256);

  const size_t SMEM = (size_t)64 * HDIM * 2 + 32 * 64 * 4 + 32 * 16 * 4; // 141312 B

  // 1) Embedding gather -> bf16 activations.
  embed_gather<<<(unsigned)(MB_ * DEMB / 256), 256, 0, stream>>>(tokens, embed, Xemb);

  // 2) Weight convert + transpose (bf16, N-major for contiguous B-frags).
  conv_transpose<<<(unsigned)((size_t)DEMB * G4 / 256), 256, 0, stream>>>(Wx0, Wx0t, DEMB, G4);
  conv_transpose<<<(unsigned)((size_t)HDIM * G4 / 256), 256, 0, stream>>>(Wh0, Wh0t, HDIM, G4);
  conv_transpose<<<(unsigned)((size_t)HDIM * PDIM / 256), 256, 0, stream>>>(Wfc0, Wf0t, HDIM, PDIM);
  conv_transpose<<<(unsigned)((size_t)PDIM * G4 / 256), 256, 0, stream>>>(Wx1, Wx1t, PDIM, G4);
  conv_transpose<<<(unsigned)((size_t)HDIM * G4 / 256), 256, 0, stream>>>(Wh1, Wh1t, HDIM, G4);
  conv_transpose<<<(unsigned)((size_t)HDIM * PDIM / 256), 256, 0, stream>>>(Wfc1, Wf1t, HDIM, PDIM);

  // ---- Layer 0 ----
  gemm_bf16<<<dim3(G4 / 32, (unsigned)(MB_ / 64)), 256, 0, stream>>>(
      Xemb, Wx0t, nullptr, nullptr, XZ, (int)MB_, G4, DEMB);
  (void)hipMemsetAsync(syncc, 0, 256, stream);
  lstm_scan<<<HDIM / 16, 256, SMEM, stream>>>(XZ, b0, Wh0t, Y, h0, h1, syncc);
  gemm_bf16<<<dim3(PDIM / 32, (unsigned)(MB_ / 64)), 256, 0, stream>>>(
      Y, Wf0t, bfc0, nullptr, X1, (int)MB_, PDIM, HDIM);

  // ---- Layer 1 ----
  gemm_bf16<<<dim3(G4 / 32, (unsigned)(MB_ / 64)), 256, 0, stream>>>(
      X1, Wx1t, nullptr, nullptr, XZ, (int)MB_, G4, PDIM);
  (void)hipMemsetAsync(syncc, 0, 256, stream);
  lstm_scan<<<HDIM / 16, 256, SMEM, stream>>>(XZ, b1, Wh1t, Y, h0, h1, syncc);
  gemm_bf16<<<dim3(PDIM / 32, (unsigned)(MB_ / 64)), 256, 0, stream>>>(
      Y, Wf1t, bfc1, (float*)d_out, nullptr, (int)MB_, PDIM, HDIM);
}
